// EIRNN_37606733643979
// MI455X (gfx1250) — compile-verified, matrix operand order checked by reading, not run on hardware
//
#include <hip/hip_runtime.h>
#include <math.h>

// ---------------------------------------------------------------------------
// EI-RNN for MI455X (gfx1250, wave32).
//  - persistent scan kernel: 64 WGs x 4 waves, one wave per 16x16 (b,h) tile
//  - recurrent + input GEMMs via v_wmma_f32_16x16x32_bf16 (fp32 accumulate)
//  - hidden state held in WMMA accumulator registers for all 512 steps
//  - B-matrix tiles staged once in LDS (36KB/WGP), read via ds_load_b128
//  - activations/x pre-packed in exact WMMA A-register bf16 layout so the
//    serial inner loop is pure {ds_load + global_load_b128 + v_wmma}
//  - agent-scope atomic grid barrier between steps (no per-step launches)
// ---------------------------------------------------------------------------

typedef __attribute__((ext_vector_type(16))) __bf16       v16bf;
typedef __attribute__((ext_vector_type(8)))  float        v8f;
typedef __attribute__((ext_vector_type(2)))  float        v2f;
typedef __attribute__((ext_vector_type(8)))  unsigned int v8u;

#define T_STEPS 512
#define BATCH   64
#define INSZ    128
#define HID     1024
#define ESZ     819
#define OSZ     64

#define NT_H    64   // HID/16 hidden n-tiles
#define KT_H    32   // HID/32 k-tiles for recurrent GEMM
#define KT_IN   4    // INSZ/32 k-tiles for input GEMM
#define KT_E    26   // ceil(ESZ/32) k-tiles for readout GEMM (pad 819->832)
#define NT_O    4    // OSZ/16
#define NWG     64   // workgroups in the persistent scan kernel

#define ASLOT_DW (4 * KT_H * 256)            // 32768 dwords = 128KB per slot

// ---------- bf16 helpers ----------------------------------------------------
__device__ __forceinline__ unsigned short f32_to_bf16_rne(float f) {
  unsigned u = __builtin_bit_cast(unsigned, f);
  unsigned r = 0x7FFFu + ((u >> 16) & 1u);
  return (unsigned short)((u + r) >> 16);
}
__device__ __forceinline__ unsigned pack2_bf16(float lo, float hi) {
  return (unsigned)f32_to_bf16_rne(lo) | ((unsigned)f32_to_bf16_rne(hi) << 16);
}
__device__ __forceinline__ __bf16 bf16_bits(unsigned short s) {
  return __builtin_bit_cast(__bf16, s);
}

// fp32 row -> bf16 A fragment (used only off the critical path, in fc_kernel)
__device__ __forceinline__ v16bf load_a_frag_f32(const float* __restrict__ row,
                                                 int kbase, int hi) {
  v16bf a;
#pragma unroll
  for (int v = 0; v < 8; ++v) {
    int k = kbase + 16 * (v >> 2) + 8 * hi + 2 * (v & 3);
    v2f f = *(const v2f*)(row + k);
    a[2 * v]     = bf16_bits(f32_to_bf16_rne(f.x));
    a[2 * v + 1] = bf16_bits(f32_to_bf16_rne(f.y));
  }
  return a;
}

// pre-packed fragment (lane-major, 8 dwords per lane, contiguous 32B)
__device__ __forceinline__ v16bf load_frag_packed(const unsigned* __restrict__ p) {
  v8u raw = *(const v8u*)p;
  return __builtin_bit_cast(v16bf, raw);
}

__device__ __forceinline__ v8f wmma_bf16(v16bf a, v16bf b, v8f c) {
  return __builtin_amdgcn_wmma_f32_16x16x32_bf16(
      /*neg_a=*/false, a, /*neg_b=*/false, b,
      /*c_mod=*/(short)0, c, /*reuse_a=*/false, /*reuse_b=*/false);
}

__device__ __forceinline__ float fast_tanh(float s) {
#if __has_builtin(__builtin_amdgcn_tanhf)
  return __builtin_amdgcn_tanhf(s);
#else
  return tanhf(s);
#endif
}

// ---------------------------------------------------------------------------
// Packing kernels. Fragment tile = 256 dwords, dword index = lane*8 + v.
// B layout (32x16 KxN): lane -> n=lane&15, hi=lane>>4;
//   VGPR v holds k = 16*(v>>2) + 8*hi + 2*(v&3) and k+1.
// A layout (16x32 MxK): lane -> m=lane&15, hi=lane>>4; same k formula.
// ---------------------------------------------------------------------------
__global__ void pack_wt_kernel(const float* __restrict__ w_hh,
                               const float* __restrict__ mask,
                               unsigned* __restrict__ wt) {
  int g    = blockIdx.x * blockDim.x + threadIdx.x;   // NT_H*KT_H*256
  int tile = g >> 8, r = g & 255;
  int lane = r >> 3, v = r & 7;
  int nt = tile >> 5, kt = tile & 31;
  int hi = lane >> 4;
  int n  = nt * 16 + (lane & 15);
  int k  = kt * 32 + 16 * (v >> 2) + 8 * hi + 2 * (v & 3);
  float v0 = fabsf(w_hh[n * HID + k])     * mask[n * HID + k];
  float v1 = fabsf(w_hh[n * HID + k + 1]) * mask[n * HID + k + 1];
  wt[g] = pack2_bf16(v0, v1);
}

__global__ void pack_wih_kernel(const float* __restrict__ w_ih,
                                unsigned* __restrict__ wih) {
  int g    = blockIdx.x * blockDim.x + threadIdx.x;   // NT_H*KT_IN*256
  int tile = g >> 8, r = g & 255;
  int lane = r >> 3, v = r & 7;
  int nt = tile >> 2, kt = tile & 3;
  int hi = lane >> 4;
  int n  = nt * 16 + (lane & 15);
  int k  = kt * 32 + 16 * (v >> 2) + 8 * hi + 2 * (v & 3);
  wih[g] = pack2_bf16(w_ih[n * INSZ + k], w_ih[n * INSZ + k + 1]);
}

__global__ void pack_wfc_kernel(const float* __restrict__ w_fc,
                                unsigned* __restrict__ wfc) {
  int g    = blockIdx.x * blockDim.x + threadIdx.x;   // NT_O*KT_E*256
  int tile = g >> 8, r = g & 255;
  int lane = r >> 3, v = r & 7;
  int nt = tile / KT_E, kt = tile % KT_E;
  int hi = lane >> 4;
  int n  = nt * 16 + (lane & 15);
  int k  = kt * 32 + 16 * (v >> 2) + 8 * hi + 2 * (v & 3);
  float v0 = (k     < ESZ) ? w_fc[n * ESZ + k]     : 0.f;
  float v1 = (k + 1 < ESZ) ? w_fc[n * ESZ + k + 1] : 0.f;
  wfc[g] = pack2_bf16(v0, v1);
}

// x -> bf16 A-fragment layout: per t, tiles (mt 0..3, kt 0..3)
__global__ void pack_x_kernel(const float* __restrict__ x,
                              unsigned* __restrict__ xp) {
  int g    = blockIdx.x * blockDim.x + threadIdx.x;   // T*16*256
  int r    = g & 255;
  int tile = (g >> 8) & 15;
  int t    = g >> 12;
  int lane = r >> 3, v = r & 7;
  int mt = tile >> 2, kt = tile & 3;
  int hi = lane >> 4, m = lane & 15;
  int k  = kt * 32 + 16 * (v >> 2) + 8 * hi + 2 * (v & 3);
  const float* row = x + ((size_t)t * BATCH + mt * 16 + m) * INSZ;
  xp[g] = pack2_bf16(row[k], row[k + 1]);
}

__global__ void init_counter_kernel(unsigned* __restrict__ c) {
  if (threadIdx.x == 0 && blockIdx.x == 0) *c = 0u;
}

// ---------------------------------------------------------------------------
// Persistent scan kernel.
// wave w of block b owns tile (mt=w in 0..3, nt=b in 0..63).
// abuf: double-buffered bf16 activations in A-fragment layout,
//       slot = ASLOT_DW dwords, tile index = mt*KT_H + kt.
// ---------------------------------------------------------------------------
__global__ void __launch_bounds__(128)
scan_kernel(const unsigned* __restrict__ xpack,
            const float* __restrict__ b_ih,
            const float* __restrict__ b_hh,
            const float* __restrict__ alpha,
            const unsigned* __restrict__ wt,
            const unsigned* __restrict__ wih,
            float* __restrict__ activity,       // (T,B,HID) fp32 = d_out tail
            unsigned* __restrict__ abuf,        // 2 * ASLOT_DW dwords
            unsigned* __restrict__ counter) {
  __shared__ unsigned lds_wt[KT_H * 256];       // 32KB: this nt's B tiles
  __shared__ unsigned lds_wih[KT_IN * 256];     // 4KB

  const int wave = threadIdx.x >> 5;
  const int lane = threadIdx.x & 31;
  const int hi   = lane >> 4, ln = lane & 15;
  const int mt   = wave;          // batch tile  (0..3)
  const int nt   = blockIdx.x;    // hidden tile (0..63)
  const int row0 = mt * 16, col0 = nt * 16;

  // stage B tiles for this nt into LDS (shared by all 4 waves)
  {
    const uint4* g0 = (const uint4*)(wt  + (size_t)nt * KT_H  * 256);
    const uint4* g1 = (const uint4*)(wih + (size_t)nt * KT_IN * 256);
    for (int i = threadIdx.x; i < (KT_H * 256) / 4; i += 128)
      ((uint4*)lds_wt)[i] = g0[i];
    for (int i = threadIdx.x; i < (KT_IN * 256) / 4; i += 128)
      ((uint4*)lds_wih)[i] = g1[i];
  }
  __syncthreads();

  const float a    = alpha[0];
  const float bias = b_ih[col0 + ln] + b_hh[col0 + ln];

  // consumer view of packed activations (this wave's mt, all kt)
  const unsigned* aslot0 = abuf + (size_t)mt * KT_H * 256 + lane * 8;
  const unsigned* aslot1 = aslot0 + ASLOT_DW;
  // producer scatter: this wave's 16x16 tile lands in A-tile kt = nt>>1.
  // element (m = v+8*hi, kk = 16*(nt&1)+ln):
  //   lane' = m + 16*((ln>>3)&1), v' = 4*(nt&1) + ((ln&7)>>1), parity = ln&1
  unsigned short* ab16 = (unsigned short*)abuf;
  const int prod_base_hw =
      (((mt * KT_H + (nt >> 1)) * 256 +
        (8 * hi + 16 * ((ln >> 3) & 1)) * 8 +
        (4 * (nt & 1) + ((ln & 7) >> 1))) << 1) + (ln & 1);

  const unsigned* xp = xpack + (size_t)mt * 4 * 256 + lane * 8;

  v8f state = {0.f, 0.f, 0.f, 0.f, 0.f, 0.f, 0.f, 0.f};

  for (int t = 0; t < T_STEPS; ++t) {
    v8f c = {0.f, 0.f, 0.f, 0.f, 0.f, 0.f, 0.f, 0.f};

    // u_t tile: packed x fragments x LDS-resident w_ih tiles (4 WMMAs)
    const unsigned* xpt = xp + (size_t)t * 16 * 256;
#pragma unroll
    for (int kt = 0; kt < KT_IN; ++kt) {
      v16bf af = load_frag_packed(xpt + kt * 256);
      v16bf bf = load_frag_packed(&lds_wih[kt * 256 + lane * 8]);
      c = wmma_bf16(af, bf, c);
    }

    // recurrent tile: packed activations x LDS-resident eff_w tiles (32 WMMAs)
    if (t > 0) {
      const unsigned* ap = ((t - 1) & 1) ? aslot1 : aslot0;
#pragma unroll 8
      for (int kt = 0; kt < KT_H; ++kt) {
        v16bf af = load_frag_packed(ap + kt * 256);
        v16bf bf = load_frag_packed(&lds_wt[kt * 256 + lane * 8]);
        c = wmma_bf16(af, bf, c);
      }
    }

    // state update + tanh; write fp32 activity (output) + bf16 packed copy
    float* orow = activity + ((size_t)t * BATCH + row0) * HID + col0 + ln;
    unsigned short* prow = ab16 + (size_t)(t & 1) * (ASLOT_DW * 2) + prod_base_hw;
#pragma unroll
    for (int v = 0; v < 8; ++v) {
      float total = c[v] + bias;
      float s = state[v];
      s = a * (total - s) + s;
      state[v] = s;
      float o = fast_tanh(s);
      orow[(size_t)(v + 8 * hi) * HID] = o;
      prow[v * 16] = f32_to_bf16_rne(o);   // (lane'+ : +8 dwords per v) * 2
    }

    // device-wide barrier between steps
    if (t + 1 < T_STEPS) {
      __syncthreads();
      if (threadIdx.x == 0) {
        __hip_atomic_fetch_add(counter, 1u, __ATOMIC_ACQ_REL,
                               __HIP_MEMORY_SCOPE_AGENT);
        const unsigned target = (unsigned)(NWG * (t + 1));
        while (__hip_atomic_load(counter, __ATOMIC_ACQUIRE,
                                 __HIP_MEMORY_SCOPE_AGENT) < target) {
          __builtin_amdgcn_s_sleep(1);
        }
      }
      __syncthreads();
      __threadfence();
    }
  }
}

// ---------------------------------------------------------------------------
// Readout: out[t,b,o] = act[t,b,:819] @ w_fc[o,:819] + b_fc[o]
// ---------------------------------------------------------------------------
__global__ void __launch_bounds__(128)
fc_kernel(const float* __restrict__ activity,
          const unsigned* __restrict__ wfc,
          const float* __restrict__ b_fc,
          float* __restrict__ out) {
  const int wave = threadIdx.x >> 5;
  const int lane = threadIdx.x & 31;
  const int hi   = lane >> 4, ln = lane & 15;
  const int mt   = blockIdx.x;    // 0..2047
  const int nt   = wave;          // 0..3
  const int row0 = mt * 16, col0 = nt * 16;

  const float* arow = activity + (size_t)(row0 + ln) * HID;
  const unsigned* wp = wfc + (size_t)nt * KT_E * 256 + lane * 8;

  v8f c = {0.f, 0.f, 0.f, 0.f, 0.f, 0.f, 0.f, 0.f};
#pragma unroll 2
  for (int kt = 0; kt < KT_E; ++kt) {
    v16bf af = load_a_frag_f32(arow, kt * 32, hi);  // cols 819..831: zero wgts
    v16bf bf = load_frag_packed(wp + kt * 256);
    c = wmma_bf16(af, bf, c);
  }

  const float bb = b_fc[col0 + ln];
#pragma unroll
  for (int v = 0; v < 8; ++v) {
    out[(size_t)(row0 + v + 8 * hi) * OSZ + col0 + ln] = c[v] + bb;
  }
}

// ---------------------------------------------------------------------------
extern "C" void kernel_launch(void* const* d_in, const int* in_sizes, int n_in,
                              void* d_out, int out_size, void* d_ws,
                              size_t ws_size, hipStream_t stream) {
  (void)in_sizes; (void)n_in; (void)out_size; (void)ws_size;

  const float* x     = (const float*)d_in[0];
  const float* w_ih  = (const float*)d_in[1];
  const float* b_ih  = (const float*)d_in[2];
  const float* w_hh  = (const float*)d_in[3];
  const float* b_hh  = (const float*)d_in[4];
  const float* w_fc  = (const float*)d_in[5];
  const float* b_fc  = (const float*)d_in[6];
  const float* alpha = (const float*)d_in[7];
  const float* mask  = (const float*)d_in[8];

  float* out      = (float*)d_out;                       // (T,B,O)
  float* activity = out + (size_t)T_STEPS * BATCH * OSZ; // (T,B,HID)

  // ws carve-up (~10.7MB):
  // counter | wt(2MB) | wih(256KB) | wfc(104KB) | abuf(256KB) | xpack(8MB)
  unsigned* counter = (unsigned*)d_ws;
  unsigned* wt    = (unsigned*)((char*)d_ws + 256);
  unsigned* wih   = wt    + (size_t)NT_H * KT_H  * 256;
  unsigned* wfc   = wih   + (size_t)NT_H * KT_IN * 256;
  unsigned* abuf  = wfc   + (size_t)NT_O * KT_E  * 256;
  unsigned* xpack = abuf  + (size_t)2 * ASLOT_DW;

  init_counter_kernel<<<1, 64, 0, stream>>>(counter);
  pack_wt_kernel <<<(NT_H * KT_H  * 256) / 256, 256, 0, stream>>>(w_hh, mask, wt);
  pack_wih_kernel<<<(NT_H * KT_IN * 256) / 256, 256, 0, stream>>>(w_ih, wih);
  pack_wfc_kernel<<<(NT_O * KT_E  * 256) / 256, 256, 0, stream>>>(w_fc, wfc);
  pack_x_kernel  <<<(T_STEPS * 16 * 256) / 256, 256, 0, stream>>>(x, xpack);

  scan_kernel<<<NWG, 128, 0, stream>>>(xpack, b_ih, b_hh, alpha, wt, wih,
                                       activity, abuf, counter);

  fc_kernel<<<(T_STEPS * BATCH) / 16, 128, 0, stream>>>(activity, wfc, b_fc,
                                                        out);
}
